// Net_420906795534
// MI455X (gfx1250) — compile-verified, matrix operand order checked by reading, not run on hardware
//
#include <hip/hip_runtime.h>
#include <hip/hip_bf16.h>

// ---------------------------------------------------------------------------
// Problem constants (fixed by the reference)
// ---------------------------------------------------------------------------
#define B_GRAPHS 64
#define N_NODES  1024
#define DEG      16
#define E_EDGES  (B_GRAPHS * N_NODES * DEG)   // 1,048,576
#define IN_F     20
#define HID      128
#define KEEP1    820     // ceil(0.8 * 1024)
#define KEEP2    656     // ceil(0.8 * 820)
#define KEEP3    525     // ceil(0.8 * 656)
#define M0T      (B_GRAPHS * N_NODES)  // 65536
#define M1T      (B_GRAPHS * KEEP1)    // 52480
#define M2T      (B_GRAPHS * KEEP2)    // 41984
#define M3T      (B_GRAPHS * KEEP3)    // 33600

typedef __attribute__((ext_vector_type(16))) __bf16 v16bf;
typedef __attribute__((ext_vector_type(8)))  __bf16 v8bf;
typedef __attribute__((ext_vector_type(8)))  float  v8f;

// ---------------------------------------------------------------------------
// Utility kernels
// ---------------------------------------------------------------------------
__global__ void zero_f32_kernel(float* __restrict__ p, int n) {
  int i = blockIdx.x * blockDim.x + threadIdx.x;
  if (i < n) p[i] = 0.0f;
}

__global__ void fill_i32_kernel(int* __restrict__ p, int n, int val) {
  int i = blockIdx.x * blockDim.x + threadIdx.x;
  if (i < n) p[i] = val;
}

__global__ void init_edges_kernel(const int* __restrict__ edge_index,
                                  int* __restrict__ src, int* __restrict__ dst,
                                  float* __restrict__ emask, int E) {
  int e = blockIdx.x * blockDim.x + threadIdx.x;
  if (e >= E) return;
  src[e]   = edge_index[e];
  dst[e]   = edge_index[E + e];
  emask[e] = 1.0f;
}

// ---------------------------------------------------------------------------
// Edge message scatter:  agg[dst] += x[src] * emask   (segment_sum).
// One thread per (edge, 4-channel group): float4 gather + 4 f32 atomics.
// The whole working set (~115 MB) is L2-resident (192 MB), so atomics
// resolve on-chip.  C is 20 or 128, both divisible by 4; 16B alignment holds
// (row strides 80B / 512B).
// ---------------------------------------------------------------------------
__global__ void scatter4_kernel(const float* __restrict__ x,
                                const int* __restrict__ src,
                                const int* __restrict__ dst,
                                const float* __restrict__ emask,
                                float* __restrict__ agg, int EG, int C4) {
  int t = blockIdx.x * blockDim.x + threadIdx.x;
  if (t >= EG) return;
  int e  = t / C4;
  int c4 = t - e * C4;
  float m = emask[e];
  if (m == 0.0f) return;
  const int C = C4 * 4;
  float4 v = *(const float4*)(x + (size_t)src[e] * C + c4 * 4);
  float* base = agg + (size_t)dst[e] * C + c4 * 4;
  atomicAdd(base + 0, v.x * m);
  atomicAdd(base + 1, v.y * m);
  atomicAdd(base + 2, v.z * m);
  atomicAdd(base + 3, v.w * m);
}

// ---------------------------------------------------------------------------
// A-fragment loader (16-bit A layout, ISA 7.12.2): per lane the 16 elements
// are two contiguous 8-float runs [k0+8*hi, +8) and [k0+16+8*hi, +8).
// Vector (2x float4 per run) when fully in-range, guarded scalar otherwise.
// ---------------------------------------------------------------------------
__device__ inline v16bf load_a_frag(const float* __restrict__ A, int row,
                                    int K, int k0, int hi) {
  v16bf a;
  const int kb0 = k0 + hi * 8;
  const int kb1 = k0 + 16 + hi * 8;
  if (kb1 + 8 <= K) {
    const float4* p0 = (const float4*)(A + (size_t)row * K + kb0);
    const float4* p1 = (const float4*)(A + (size_t)row * K + kb1);
    float4 x0 = p0[0], x1 = p0[1], y0 = p1[0], y1 = p1[1];
    a[0] = (__bf16)x0.x;  a[1] = (__bf16)x0.y;  a[2]  = (__bf16)x0.z;  a[3]  = (__bf16)x0.w;
    a[4] = (__bf16)x1.x;  a[5] = (__bf16)x1.y;  a[6]  = (__bf16)x1.z;  a[7]  = (__bf16)x1.w;
    a[8] = (__bf16)y0.x;  a[9] = (__bf16)y0.y;  a[10] = (__bf16)y0.z;  a[11] = (__bf16)y0.w;
    a[12]= (__bf16)y1.x;  a[13]= (__bf16)y1.y;  a[14] = (__bf16)y1.z;  a[15] = (__bf16)y1.w;
  } else {
    #pragma unroll
    for (int i = 0; i < 8; ++i) {
      int ka = kb0 + i, kb = kb1 + i;
      a[i]     = (__bf16)((ka < K) ? A[(size_t)row * K + ka] : 0.0f);
      a[8 + i] = (__bf16)((kb < K) ? A[(size_t)row * K + kb] : 0.0f);
    }
  }
  return a;
}

// B-fragment from LDS (weights stored [col][k] bf16): two ds_load_b128 runs.
__device__ inline v16bf load_b_frag(const __bf16* Wl, int col, int K,
                                    int k0, int hi) {
  v16bf b;
  const int kb0 = k0 + hi * 8;
  const int kb1 = k0 + 16 + hi * 8;
  if (kb1 + 8 <= K) {
    v8bf r0 = *(const v8bf*)(Wl + col * K + kb0);
    v8bf r1 = *(const v8bf*)(Wl + col * K + kb1);
    #pragma unroll
    for (int i = 0; i < 8; ++i) { b[i] = r0[i]; b[8 + i] = r1[i]; }
  } else {
    #pragma unroll
    for (int i = 0; i < 8; ++i) {
      int ka = kb0 + i, kb = kb1 + i;
      b[i]     = (ka < K) ? Wl[col * K + ka] : (__bf16)0.0f;
      b[8 + i] = (kb < K) ? Wl[col * K + kb] : (__bf16)0.0f;
    }
  }
  return b;
}

// ---------------------------------------------------------------------------
// Fused GraphConv GEMM:  out = relu( agg @ Wrel + x @ Wroot + bias )
// N = 128, K = 20 or 128.  Block = 256 threads = 8 waves; both weight
// matrices staged once in LDS as bf16 [col][k] (<= 64 KB of the 320 KB WGP
// LDS).  Each wave owns a 16-row strip and sweeps all 8 N-tiles, reusing its
// A fragment 8x per k-step.  v_wmma_f32_16x16x32_bf16, f32 accumulation,
// wave32 with EXEC all ones as WMMA requires.  M is always a multiple of 128.
// ---------------------------------------------------------------------------
__global__ __launch_bounds__(256)
void conv_wmma_kernel(const float* __restrict__ Aagg,
                      const float* __restrict__ Aroot,
                      const float* __restrict__ Wrel,
                      const float* __restrict__ Wroot,
                      const float* __restrict__ bias,
                      float* __restrict__ out, int M, int K) {
  const int N = HID;
  __shared__ __bf16 Wl[2][HID * HID];    // [pass][col*K + k], K <= 128

  // ---- stage both weight matrices into LDS, transposed to [col][k] ----
  for (int pass = 0; pass < 2; ++pass) {
    const float* W = pass ? Wroot : Wrel;
    for (int idx = threadIdx.x; idx < N * K; idx += 256) {
      int col = idx / K;
      int k   = idx - col * K;
      Wl[pass][idx] = (__bf16)W[k * N + col];
    }
  }
  __syncthreads();

  const int wid    = threadIdx.x >> 5;   // wave 0..7
  const int lane   = threadIdx.x & 31;
  const int laneLo = lane & 15;
  const int hi     = (lane >> 4) & 1;
  const int rowBase = blockIdx.x * 128 + wid * 16;
  const int rowA    = rowBase + laneLo;

  v8f acc[8] = {};

  #pragma unroll
  for (int pass = 0; pass < 2; ++pass) {
    const float* A = pass ? Aroot : Aagg;
    for (int k0 = 0; k0 < K; k0 += 32) {
      v16bf a = load_a_frag(A, rowA, K, k0, hi);
      #pragma unroll
      for (int tn = 0; tn < 8; ++tn) {
        v16bf b = load_b_frag(&Wl[pass][0], tn * 16 + laneLo, K, k0, hi);
        acc[tn] = __builtin_amdgcn_wmma_f32_16x16x32_bf16(
                      /*neg_a=*/false, a, /*neg_b=*/false, b,
                      /*c_mod=*/(short)0, acc[tn],
                      /*reuse_a=*/false, /*reuse_b=*/false);
      }
    }
  }

  // C/D layout: VGPR r -> (M = r + 8*hi, N = laneLo)
  #pragma unroll
  for (int tn = 0; tn < 8; ++tn) {
    int col = tn * 16 + laneLo;
    float bv = bias[col];
    #pragma unroll
    for (int r = 0; r < 8; ++r) {
      int row = rowBase + r + hi * 8;
      float v = acc[tn][r] + bv;
      out[(size_t)row * N + col] = v > 0.0f ? v : 0.0f;
    }
  }
}

// ---------------------------------------------------------------------------
// score[i] = tanh( (h[i] . w) / ||w|| )
// ---------------------------------------------------------------------------
__global__ void score_kernel(const float* __restrict__ h,
                             const float* __restrict__ w,
                             float* __restrict__ score, int M, int C) {
  int i = blockIdx.x * blockDim.x + threadIdx.x;
  if (i >= M) return;
  float dot = 0.0f, nrm = 0.0f;
  for (int c = 0; c < C; ++c) {
    float wv = w[c];
    dot += h[i * C + c] * wv;
    nrm += wv * wv;
  }
  score[i] = tanhf(dot * rsqrtf(nrm));
}

// ---------------------------------------------------------------------------
// Per-graph top-k via bitonic sort (descending) of 1024 padded slots in LDS.
// Only the kept SET matters (readout is order-invariant, conv is remapped
// consistently), so the sort order is free.
// ---------------------------------------------------------------------------
__global__ __launch_bounds__(1024)
void topk_kernel(const float* __restrict__ score, int n, int k,
                 int* __restrict__ perm, float* __restrict__ kscore,
                 int* __restrict__ node_map) {
  __shared__ float s[1024];
  __shared__ int   id[1024];
  int g = blockIdx.x;
  int t = threadIdx.x;
  s[t]  = (t < n) ? score[g * n + t] : -1.0e30f;
  id[t] = t;
  __syncthreads();
  for (int kk = 2; kk <= 1024; kk <<= 1) {
    for (int j = kk >> 1; j > 0; j >>= 1) {
      int ixj = t ^ j;
      if (ixj > t) {
        bool desc = ((t & kk) == 0);
        bool sw = desc ? (s[t] < s[ixj]) : (s[t] > s[ixj]);
        if (sw) {
          float tf = s[t]; s[t] = s[ixj]; s[ixj] = tf;
          int   ti = id[t]; id[t] = id[ixj]; id[ixj] = ti;
        }
      }
      __syncthreads();
    }
  }
  if (t < k) {
    perm[g * k + t]         = g * n + id[t];
    kscore[g * k + t]       = s[t];
    node_map[g * n + id[t]] = g * k + t;
  }
}

// out[r, c4] = h[perm[r], c4] * kscore[r]   (float4, C = 128)
__global__ void pool_apply4_kernel(const float* __restrict__ h,
                                   const int* __restrict__ perm,
                                   const float* __restrict__ ks,
                                   float* __restrict__ out, int MC4, int C4) {
  int t = blockIdx.x * blockDim.x + threadIdx.x;
  if (t >= MC4) return;
  int r  = t / C4;
  int c4 = t - r * C4;
  const int C = C4 * 4;
  float m = ks[r];
  float4 v = *(const float4*)(h + (size_t)perm[r] * C + c4 * 4);
  v.x *= m; v.y *= m; v.z *= m; v.w *= m;
  *(float4*)(out + (size_t)r * C + c4 * 4) = v;
}

// mask-based edge reindex (like the reference: mask instead of filter)
__global__ void remap_kernel(int* __restrict__ src, int* __restrict__ dst,
                             float* __restrict__ emask,
                             const int* __restrict__ node_map, int E) {
  int e = blockIdx.x * blockDim.x + threadIdx.x;
  if (e >= E) return;
  int ns = node_map[src[e]];
  int nd = node_map[dst[e]];
  float m = emask[e];
  if (ns < 0 || nd < 0) m = 0.0f;
  src[e]   = ns < 0 ? 0 : ns;
  dst[e]   = nd < 0 ? 0 : nd;
  emask[e] = m;
}

// z[g, 0:128]   += max over rows,  z[g, 128:256] += mean over rows
__global__ __launch_bounds__(HID)
void readout_kernel(const float* __restrict__ x, float* __restrict__ z, int k) {
  int g = blockIdx.x;
  int c = threadIdx.x;                         // 128 threads
  const float* base = x + (size_t)g * k * HID + c;
  float mx = -1.0e30f, sm = 0.0f;
  for (int j = 0; j < k; ++j) {
    float v = base[(size_t)j * HID];
    mx = fmaxf(mx, v);
    sm += v;
  }
  z[g * 2 * HID + c]       += mx;
  z[g * 2 * HID + HID + c] += sm / (float)k;
}

// tiny MLP head: 256 -> 128 relu -> 64 relu -> 1 (one block per graph)
__global__ __launch_bounds__(256)
void mlp_kernel(const float* __restrict__ z,
                const float* __restrict__ W1, const float* __restrict__ b1,
                const float* __restrict__ W2, const float* __restrict__ b2,
                const float* __restrict__ W3, const float* __restrict__ b3,
                float* __restrict__ out) {
  __shared__ float zi[256];
  __shared__ float h1[128];
  __shared__ float h2[64];
  int g = blockIdx.x;
  int t = threadIdx.x;
  zi[t] = z[g * 256 + t];
  __syncthreads();
  if (t < 128) {
    float acc = b1[t];
    for (int i = 0; i < 256; ++i) acc += zi[i] * W1[i * 128 + t];
    h1[t] = acc > 0.0f ? acc : 0.0f;
  }
  __syncthreads();
  if (t < 64) {
    float acc = b2[t];
    for (int i = 0; i < 128; ++i) acc += h1[i] * W2[i * 64 + t];
    h2[t] = acc > 0.0f ? acc : 0.0f;
  }
  __syncthreads();
  if (t == 0) {
    float acc = b3[0];
    for (int i = 0; i < 64; ++i) acc += h2[i] * W3[i];
    out[g] = acc;
  }
}

// ---------------------------------------------------------------------------
// Host orchestration
// ---------------------------------------------------------------------------
static inline int cdiv(int a, int b) { return (a + b - 1) / b; }

extern "C" void kernel_launch(void* const* d_in, const int* in_sizes, int n_in,
                              void* d_out, int out_size, void* d_ws, size_t ws_size,
                              hipStream_t stream) {
  (void)in_sizes; (void)n_in; (void)out_size; (void)ws_size;

  const float* x          = (const float*)d_in[0];   // [65536, 20]
  const int*   edge_index = (const int*)d_in[1];     // [2, E]
  // d_in[2] = batch (unused: graphs are contiguous equal-size)
  const float* W_rel1 = (const float*)d_in[3];
  const float* b_rel1 = (const float*)d_in[4];
  const float* W_root1= (const float*)d_in[5];
  const float* w_pool1= (const float*)d_in[6];
  const float* W_rel2 = (const float*)d_in[7];
  const float* b_rel2 = (const float*)d_in[8];
  const float* W_root2= (const float*)d_in[9];
  const float* w_pool2= (const float*)d_in[10];
  const float* W_rel3 = (const float*)d_in[11];
  const float* b_rel3 = (const float*)d_in[12];
  const float* W_root3= (const float*)d_in[13];
  const float* w_pool3= (const float*)d_in[14];
  const float* W_lin1 = (const float*)d_in[15];
  const float* b_lin1 = (const float*)d_in[16];
  const float* W_lin2 = (const float*)d_in[17];
  const float* b_lin2 = (const float*)d_in[18];
  const float* W_lin3 = (const float*)d_in[19];
  const float* b_lin3 = (const float*)d_in[20];
  float* out = (float*)d_out;                        // [64]

  // ---- workspace layout (~115 MB) ----
  float* fA     = (float*)d_ws;            // node features (ping)  65536*128
  float* fAgg   = fA     + M0T * HID;      // segment-sum buffer    65536*128
  float* fB     = fAgg   + M0T * HID;      // conv output (pong)    65536*128
  float* fScore = fB     + M0T * HID;      // 65536
  float* fKScr  = fScore + M0T;            // 65536
  float* fZ     = fKScr  + M0T;            // 64*256
  float* fMask  = fZ     + B_GRAPHS * 256; // E
  int*   iPerm  = (int*)(fMask + E_EDGES); // 65536
  int*   iMap   = iPerm  + M0T;            // 65536
  int*   iSrc   = iMap   + M0T;            // E
  int*   iDst   = iSrc   + E_EDGES;        // E

  const int TB = 256;

  // ---- init ----
  init_edges_kernel<<<cdiv(E_EDGES, TB), TB, 0, stream>>>(edge_index, iSrc, iDst, fMask, E_EDGES);
  zero_f32_kernel<<<cdiv(B_GRAPHS * 256, TB), TB, 0, stream>>>(fZ, B_GRAPHS * 256);

  // ================= Layer 1 (M=65536, K=20) =================
  zero_f32_kernel<<<cdiv(M0T * IN_F, TB), TB, 0, stream>>>(fAgg, M0T * IN_F);
  scatter4_kernel<<<cdiv(E_EDGES * (IN_F / 4), TB), TB, 0, stream>>>(x, iSrc, iDst, fMask, fAgg, E_EDGES * (IN_F / 4), IN_F / 4);
  conv_wmma_kernel<<<M0T / 128, 256, 0, stream>>>(fAgg, x, W_rel1, W_root1, b_rel1, fB, M0T, IN_F);
  score_kernel<<<cdiv(M0T, TB), TB, 0, stream>>>(fB, w_pool1, fScore, M0T, HID);
  fill_i32_kernel<<<cdiv(M0T, TB), TB, 0, stream>>>(iMap, M0T, -1);
  topk_kernel<<<B_GRAPHS, 1024, 0, stream>>>(fScore, N_NODES, KEEP1, iPerm, fKScr, iMap);
  pool_apply4_kernel<<<cdiv(M1T * (HID / 4), TB), TB, 0, stream>>>(fB, iPerm, fKScr, fA, M1T * (HID / 4), HID / 4);
  remap_kernel<<<cdiv(E_EDGES, TB), TB, 0, stream>>>(iSrc, iDst, fMask, iMap, E_EDGES);
  readout_kernel<<<B_GRAPHS, HID, 0, stream>>>(fA, fZ, KEEP1);

  // ================= Layer 2 (M=52480, K=128) =================
  zero_f32_kernel<<<cdiv(M1T * HID, TB), TB, 0, stream>>>(fAgg, M1T * HID);
  scatter4_kernel<<<cdiv(E_EDGES * (HID / 4), TB), TB, 0, stream>>>(fA, iSrc, iDst, fMask, fAgg, E_EDGES * (HID / 4), HID / 4);
  conv_wmma_kernel<<<M1T / 128, 256, 0, stream>>>(fAgg, fA, W_rel2, W_root2, b_rel2, fB, M1T, HID);
  score_kernel<<<cdiv(M1T, TB), TB, 0, stream>>>(fB, w_pool2, fScore, M1T, HID);
  fill_i32_kernel<<<cdiv(M1T, TB), TB, 0, stream>>>(iMap, M1T, -1);
  topk_kernel<<<B_GRAPHS, 1024, 0, stream>>>(fScore, KEEP1, KEEP2, iPerm, fKScr, iMap);
  pool_apply4_kernel<<<cdiv(M2T * (HID / 4), TB), TB, 0, stream>>>(fB, iPerm, fKScr, fA, M2T * (HID / 4), HID / 4);
  remap_kernel<<<cdiv(E_EDGES, TB), TB, 0, stream>>>(iSrc, iDst, fMask, iMap, E_EDGES);
  readout_kernel<<<B_GRAPHS, HID, 0, stream>>>(fA, fZ, KEEP2);

  // ================= Layer 3 (M=41984, K=128) =================
  zero_f32_kernel<<<cdiv(M2T * HID, TB), TB, 0, stream>>>(fAgg, M2T * HID);
  scatter4_kernel<<<cdiv(E_EDGES * (HID / 4), TB), TB, 0, stream>>>(fA, iSrc, iDst, fMask, fAgg, E_EDGES * (HID / 4), HID / 4);
  conv_wmma_kernel<<<M2T / 128, 256, 0, stream>>>(fAgg, fA, W_rel3, W_root3, b_rel3, fB, M2T, HID);
  score_kernel<<<cdiv(M2T, TB), TB, 0, stream>>>(fB, w_pool3, fScore, M2T, HID);
  fill_i32_kernel<<<cdiv(M2T, TB), TB, 0, stream>>>(iMap, M2T, -1);
  topk_kernel<<<B_GRAPHS, 1024, 0, stream>>>(fScore, KEEP2, KEEP3, iPerm, fKScr, iMap);
  pool_apply4_kernel<<<cdiv(M3T * (HID / 4), TB), TB, 0, stream>>>(fB, iPerm, fKScr, fA, M3T * (HID / 4), HID / 4);
  // (no further conv: edge remap not needed after last pool)
  readout_kernel<<<B_GRAPHS, HID, 0, stream>>>(fA, fZ, KEEP3);

  // ================= MLP head =================
  mlp_kernel<<<B_GRAPHS, 256, 0, stream>>>(fZ, W_lin1, b_lin1, W_lin2, b_lin2, W_lin3, b_lin3, out);
}
